// BipartiteRGAT_26302379720716
// MI455X (gfx1250) — matrix operand order repaired and checked by reading, not compiled, not run on hardware
//
#include <hip/hip_runtime.h>

typedef _Float16 v16h __attribute__((ext_vector_type(16)));
typedef _Float16 v8h  __attribute__((ext_vector_type(8)));
typedef _Float16 v4h  __attribute__((ext_vector_type(4)));
typedef float    v8f  __attribute__((ext_vector_type(8)));

constexpr int N_VIEWS    = 3;
constexpr int NN         = 20000;
constexpr int NT         = 60000;       // N_VIEWS * NN
constexpr int E_EDGES    = 800000;
constexpr int R_REL      = 6;
constexpr int B_BASES    = 4;
constexpr int D_DIM      = 128;
constexpr int IN_FEAT    = 2000;
constexpr int NUM_LABELS = 5;
constexpr float NEG_SLOPE = 0.2f;

#define LDT 40   // LDS row stride in halves (32 data + 8 pad) -> 80B rows, 16B aligned

// ---------------------------------------------------------------------------
// Tiled WMMA GEMM: C[M x 128] = A[M x K] * B[K x 128] (+bias)(+relu)
// Block = 256 threads = 8 waves; each block does a 128x128 tile; each wave a
// 16x128 strip (8 x v8f accumulators). f32 -> f16 conversion during staging.
// blockIdx.y selects a "z" slice of B/C (used for the 6 relation GEMMs).
// ---------------------------------------------------------------------------
__global__ __launch_bounds__(256)
void wmma_gemm_n128(const float* __restrict__ A, int lda,
                    const float* __restrict__ Bm, long strideBz,
                    float* __restrict__ C, long strideCz,
                    int M, int K,
                    const float* __restrict__ bias, int doRelu)
{
    __shared__ _Float16 sX[128 * LDT];
    __shared__ _Float16 sW[128 * LDT];

    const int tid   = threadIdx.x;
    const int wave  = tid >> 5;
    const int lane  = tid & 31;
    const int mbase = blockIdx.x * 128;
    const int z     = blockIdx.y;
    Bm += (long)z * strideBz;
    C  += (long)z * strideCz;

    v8f acc[8] = {};

    for (int k0 = 0; k0 < K; k0 += 32) {
        // ---- stage A tile (128 rows x 32 K) as f16, row-major ----
        #pragma unroll
        for (int i = 0; i < 4; ++i) {
            int c   = tid + 256 * i;       // 1024 float4 chunks
            int row = c >> 3;
            int kc  = (c & 7) * 4;
            v4h hv = {};
            int grow = mbase + row;
            if (grow < M && (k0 + kc + 4) <= K) {
                const float4 f = *(const float4*)(A + (long)grow * lda + k0 + kc);
                hv[0] = (_Float16)f.x; hv[1] = (_Float16)f.y;
                hv[2] = (_Float16)f.z; hv[3] = (_Float16)f.w;
            }
            *(v4h*)(&sX[row * LDT + kc]) = hv;
        }
        // ---- stage B tile (32 K x 128 N) transposed -> sW[n][k] as f16 ----
        #pragma unroll
        for (int i = 0; i < 4; ++i) {
            int c  = tid + 256 * i;        // 1024 float4 chunks
            int kr = c >> 5;
            int n4 = (c & 31) * 4;
            float4 f = make_float4(0.f, 0.f, 0.f, 0.f);
            if (k0 + kr < K) f = *(const float4*)(Bm + (long)(k0 + kr) * 128 + n4);
            sW[(n4 + 0) * LDT + kr] = (_Float16)f.x;
            sW[(n4 + 1) * LDT + kr] = (_Float16)f.y;
            sW[(n4 + 2) * LDT + kr] = (_Float16)f.z;
            sW[(n4 + 3) * LDT + kr] = (_Float16)f.w;
        }
        __syncthreads();

        // gfx1250 global_prefetch of next A K-slice
        if (k0 + 32 < K) {
            int grow = mbase + (tid >> 1);
            if (grow < M)
                __builtin_prefetch(A + (long)grow * lda + k0 + 32 + (tid & 1) * 16, 0, 3);
        }

        // ---- A fragment: wave's 16-row strip ----
        const int arow = wave * 16 + (lane & 15);
        const int ak0  = (lane < 16) ? 0 : 8;   // K 0-7/16-23 vs 8-15/24-31
        v8h alo = *(const v8h*)(&sX[arow * LDT + ak0]);
        v8h ahi = *(const v8h*)(&sX[arow * LDT + ak0 + 16]);
        v16h afrag = __builtin_shufflevector(alo, ahi,
                        0,1,2,3,4,5,6,7,8,9,10,11,12,13,14,15);

        #pragma unroll
        for (int j = 0; j < 8; ++j) {
            const int bn = j * 16 + (lane & 15);
            const int bk = (lane < 16) ? 0 : 8;   // K 0-7/16-23 vs 8-15/24-31
            v8h blo = *(const v8h*)(&sW[bn * LDT + bk]);
            v8h bhi = *(const v8h*)(&sW[bn * LDT + bk + 16]);
            v16h bfrag = __builtin_shufflevector(blo, bhi,
                            0,1,2,3,4,5,6,7,8,9,10,11,12,13,14,15);
            acc[j] = __builtin_amdgcn_wmma_f32_16x16x32_f16(
                        false, afrag, false, bfrag, (short)0, acc[j], false, false);
        }
        __syncthreads();
    }

    // ---- epilogue: C layout VGPR r -> (M = r or r+8, N = lane%16) ----
    #pragma unroll
    for (int j = 0; j < 8; ++j) {
        const int col = j * 16 + (lane & 15);
        const float bv = bias ? bias[col] : 0.0f;
        #pragma unroll
        for (int r = 0; r < 8; ++r) {
            int row = mbase + wave * 16 + ((lane < 16) ? r : r + 8);
            if (row < M) {
                float v = acc[j][r] + bv;
                if (doRelu) v = fmaxf(v, 0.0f);
                C[(long)row * 128 + col] = v;
            }
        }
    }
}

// ---------------------------------------------------------------------------
// w[r,i,o] = sum_b comp[r,b] * basis[b,i,o]
// ---------------------------------------------------------------------------
__global__ void compute_wrel(const float* __restrict__ comp,
                             const float* __restrict__ basis,
                             float* __restrict__ wrel)
{
    int idx = blockIdx.x * 256 + threadIdx.x;
    if (idx >= R_REL * D_DIM * D_DIM) return;
    int r  = idx / (D_DIM * D_DIM);
    int io = idx % (D_DIM * D_DIM);
    float s = 0.f;
    #pragma unroll
    for (int b = 0; b < B_BASES; ++b)
        s += comp[r * B_BASES + b] * basis[(long)b * D_DIM * D_DIM + io];
    wrel[idx] = s;
}

// wq[r,i] = sum_o wrel[r,i,o]*att_q[o];  wk likewise (H==1)
__global__ void compute_wqk(const float* __restrict__ wrel,
                            const float* __restrict__ attq,
                            const float* __restrict__ attk,
                            float* __restrict__ wq, float* __restrict__ wk)
{
    int idx = blockIdx.x * 256 + threadIdx.x;
    if (idx >= R_REL * D_DIM) return;
    const float* row = wrel + (long)idx * D_DIM;
    float sq = 0.f, sk = 0.f;
    for (int o = 0; o < D_DIM; ++o) {
        float w = row[o];
        sq += w * attq[o];
        sk += w * attk[o];
    }
    wq[idx] = sq; wk[idx] = sk;
}

// q[r,n] = h[n] . wq[r]  (wave per node, lane-parallel over D, shfl reduce)
__global__ __launch_bounds__(256)
void compute_qk(const float* __restrict__ h,
                const float* __restrict__ wq, const float* __restrict__ wk,
                float* __restrict__ q, float* __restrict__ k)
{
    int wave = threadIdx.x >> 5, lane = threadIdx.x & 31;
    int node = blockIdx.x * 8 + wave;
    if (node >= NT) return;
    float4 hv = *(const float4*)(h + (long)node * 128 + lane * 4);
    float qa[R_REL], ka[R_REL];
    #pragma unroll
    for (int r = 0; r < R_REL; ++r) {
        float4 wqv = *(const float4*)(wq + r * 128 + lane * 4);
        float4 wkv = *(const float4*)(wk + r * 128 + lane * 4);
        qa[r] = hv.x*wqv.x + hv.y*wqv.y + hv.z*wqv.z + hv.w*wqv.w;
        ka[r] = hv.x*wkv.x + hv.y*wkv.y + hv.z*wkv.z + hv.w*wkv.w;
    }
    #pragma unroll
    for (int r = 0; r < R_REL; ++r) {
        float vq = qa[r], vk = ka[r];
        for (int m = 16; m > 0; m >>= 1) {
            vq += __shfl_xor(vq, m, 32);
            vk += __shfl_xor(vk, m, 32);
        }
        if (lane == 0) { q[(long)r * NT + node] = vq; k[(long)r * NT + node] = vk; }
    }
}

// order-preserving float<->uint mapping for atomic max
__device__ __forceinline__ unsigned enc_f32(float f) {
    unsigned u = __float_as_uint(f);
    return (u & 0x80000000u) ? ~u : (u | 0x80000000u);
}
__device__ __forceinline__ float dec_f32(unsigned u) {
    return __uint_as_float((u & 0x80000000u) ? (u ^ 0x80000000u) : ~u);
}

__global__ void init_nodes(unsigned* __restrict__ amaxu, float* __restrict__ denom)
{
    int i = blockIdx.x * 256 + threadIdx.x;
    if (i < NT) { amaxu[i] = 0x007FFFFFu; /* enc(-inf) */ denom[i] = 0.f; }
}

__global__ void zero_f32(float* __restrict__ p, long n)
{
    long i = (long)blockIdx.x * 256 + threadIdx.x;
    if (i < n) p[i] = 0.f;
}

// pass 1: alpha = leaky_relu(q[t,dst] + k[t,src]); segment max via atomicMax
__global__ void edge_alpha_max(const int* __restrict__ et, const int* __restrict__ es,
                               const int* __restrict__ ed,
                               const float* __restrict__ q, const float* __restrict__ k,
                               float* __restrict__ alpha, unsigned* __restrict__ amaxu)
{
    int e = blockIdx.x * 256 + threadIdx.x;
    if (e >= E_EDGES) return;
    int t = et[e], s = es[e], d = ed[e];
    float a = q[(long)t * NT + d] + k[(long)t * NT + s];
    a = (a > 0.f) ? a : NEG_SLOPE * a;
    alpha[e] = a;
    atomicMax(&amaxu[d], enc_f32(a));
}

// pass 2: ealpha = exp(alpha - amax[dst]); denom via atomicAdd
__global__ void edge_exp_sum(const int* __restrict__ ed,
                             const float* __restrict__ alpha,
                             const unsigned* __restrict__ amaxu,
                             float* __restrict__ ealpha, float* __restrict__ denom)
{
    int e = blockIdx.x * 256 + threadIdx.x;
    if (e >= E_EDGES) return;
    int d = ed[e];
    float ea = __expf(alpha[e] - dec_f32(amaxu[d]));
    ealpha[e] = ea;
    atomicAdd(&denom[d], ea);
}

// pass 3: agg[dst] += coef * xw[type, src]; wave per edge, float4 per lane
__global__ __launch_bounds__(256)
void edge_aggregate(const int* __restrict__ et, const int* __restrict__ es,
                    const int* __restrict__ ed,
                    const float* __restrict__ ealpha, const float* __restrict__ denom,
                    const float* __restrict__ xw, float* __restrict__ agg)
{
    int wave = threadIdx.x >> 5, lane = threadIdx.x & 31;
    long e = (long)blockIdx.x * 8 + wave;
    if (e >= E_EDGES) return;
    int t = et[e], s = es[e], d = ed[e];
    float coef = ealpha[e] / (denom[d] + 1e-16f);
    float4 xv = *(const float4*)(xw + ((long)t * NT + s) * 128 + lane * 4);
    float* o = agg + (long)d * 128 + lane * 4;
    atomicAdd(o + 0, coef * xv.x);
    atomicAdd(o + 1, coef * xv.y);
    atomicAdd(o + 2, coef * xv.z);
    atomicAdd(o + 3, coef * xv.w);
}

// relu(agg + conv_bias), gather views, feat @ w_int + b_int; wave per node
__global__ __launch_bounds__(256)
void finalize(const float* __restrict__ agg, const float* __restrict__ convb,
              const float* __restrict__ wint, const float* __restrict__ bint,
              float* __restrict__ out)
{
    int wave = threadIdx.x >> 5, lane = threadIdx.x & 31;
    int n = blockIdx.x * 8 + wave;
    if (n >= NN) return;
    float acc[NUM_LABELS] = {0.f, 0.f, 0.f, 0.f, 0.f};
    #pragma unroll
    for (int i = 0; i < 12; ++i) {          // 384 = 32 lanes * 12
        int idx = lane + 32 * i;            // = v*128 + d  (views concatenated)
        int v = idx >> 7, d = idx & 127;
        float val = agg[((long)v * NN + n) * 128 + d] + convb[d];
        val = fmaxf(val, 0.f);
        #pragma unroll
        for (int l = 0; l < NUM_LABELS; ++l)
            acc[l] += val * wint[idx * NUM_LABELS + l];
    }
    #pragma unroll
    for (int l = 0; l < NUM_LABELS; ++l) {
        float v = acc[l];
        for (int m = 16; m > 0; m >>= 1) v += __shfl_xor(v, m, 32);
        if (lane == 0) out[(long)n * NUM_LABELS + l] = v + bint[l];
    }
}

// ---------------------------------------------------------------------------
extern "C" void kernel_launch(void* const* d_in, const int* in_sizes, int n_in,
                              void* d_out, int out_size, void* d_ws, size_t ws_size,
                              hipStream_t stream)
{
    (void)in_sizes; (void)n_in; (void)out_size; (void)ws_size;
    // setup_inputs() dict order:
    const float* x[3]  = { (const float*)d_in[0], (const float*)d_in[3], (const float*)d_in[6] };
    const float* wp[3] = { (const float*)d_in[1], (const float*)d_in[4], (const float*)d_in[7] };
    const float* bp[3] = { (const float*)d_in[2], (const float*)d_in[5], (const float*)d_in[8] };
    const int* esrc  = (const int*)d_in[9];
    const int* edst  = (const int*)d_in[10];
    const int* etype = (const int*)d_in[11];
    const float* basis = (const float*)d_in[12];
    const float* comp  = (const float*)d_in[13];
    const float* attq  = (const float*)d_in[14];
    const float* attk  = (const float*)d_in[15];
    const float* convb = (const float*)d_in[16];
    const float* wint  = (const float*)d_in[17];
    const float* bint  = (const float*)d_in[18];
    float* out = (float*)d_out;

    // workspace carve-out (256B aligned)
    char* base = (char*)d_ws;
    size_t off = 0;
    auto alloc = [&](size_t bytes) -> void* {
        off = (off + 255) & ~(size_t)255;
        void* p = base + off;
        off += bytes;
        return p;
    };
    float*    h      = (float*)alloc((size_t)NT * 128 * 4);
    float*    wrel   = (float*)alloc((size_t)R_REL * D_DIM * D_DIM * 4);
    float*    wq     = (float*)alloc((size_t)R_REL * D_DIM * 4);
    float*    wk     = (float*)alloc((size_t)R_REL * D_DIM * 4);
    float*    xw     = (float*)alloc((size_t)R_REL * NT * 128 * 4);
    float*    qn     = (float*)alloc((size_t)R_REL * NT * 4);
    float*    kn     = (float*)alloc((size_t)R_REL * NT * 4);
    float*    alpha  = (float*)alloc((size_t)E_EDGES * 4);
    float*    ealpha = (float*)alloc((size_t)E_EDGES * 4);
    unsigned* amaxu  = (unsigned*)alloc((size_t)NT * 4);
    float*    denom  = (float*)alloc((size_t)NT * 4);
    float*    agg    = h;   // alias: h is dead after compute_qk

    // 1) per-view projection: h[v*NN : (v+1)*NN] = relu(x_v @ w_v + b_v)
    for (int v = 0; v < 3; ++v) {
        hipLaunchKernelGGL(wmma_gemm_n128, dim3((NN + 127) / 128, 1), dim3(256), 0, stream,
                           x[v], IN_FEAT, wp[v], 0L,
                           h + (long)v * NN * 128, 0L,
                           NN, IN_FEAT, bp[v], 1);
    }
    // 2) relation weights + fused attention vectors
    hipLaunchKernelGGL(compute_wrel, dim3((R_REL * D_DIM * D_DIM + 255) / 256), dim3(256),
                       0, stream, comp, basis, wrel);
    hipLaunchKernelGGL(compute_wqk, dim3((R_REL * D_DIM + 255) / 256), dim3(256),
                       0, stream, wrel, attq, attk, wq, wk);
    // 3) xw[r] = h @ w_r  (6 relation slices via blockIdx.y)
    hipLaunchKernelGGL(wmma_gemm_n128, dim3((NT + 127) / 128, R_REL), dim3(256), 0, stream,
                       h, 128, wrel, (long)D_DIM * D_DIM,
                       xw, (long)NT * 128,
                       NT, 128, (const float*)nullptr, 0);
    // 4) per-(relation,node) attention logits
    hipLaunchKernelGGL(compute_qk, dim3((NT + 7) / 8), dim3(256), 0, stream,
                       h, wq, wk, qn, kn);
    // 5) softmax over incoming edges + weighted scatter aggregation
    hipLaunchKernelGGL(init_nodes, dim3((NT + 255) / 256), dim3(256), 0, stream, amaxu, denom);
    hipLaunchKernelGGL(zero_f32, dim3((int)(((long)NT * 128 + 255) / 256)), dim3(256), 0, stream,
                       agg, (long)NT * 128);
    hipLaunchKernelGGL(edge_alpha_max, dim3((E_EDGES + 255) / 256), dim3(256), 0, stream,
                       etype, esrc, edst, qn, kn, alpha, amaxu);
    hipLaunchKernelGGL(edge_exp_sum, dim3((E_EDGES + 255) / 256), dim3(256), 0, stream,
                       edst, alpha, amaxu, ealpha, denom);
    hipLaunchKernelGGL(edge_aggregate, dim3((E_EDGES + 7) / 8), dim3(256), 0, stream,
                       etype, esrc, edst, ealpha, denom, xw, agg);
    // 6) relu + view gather + final linear to labels
    hipLaunchKernelGGL(finalize, dim3((NN + 7) / 8), dim3(256), 0, stream,
                       agg, convb, wint, bint, out);
}
